// SparseMoEEngine_46359876993227
// MI455X (gfx1250) — matrix operand order repaired and checked by reading, not compiled, not run on hardware
//
#include <hip/hip_runtime.h>
#include <math.h>

// Problem constants (match reference setup_inputs)
#define T_TOK 4096
#define D_DIM 2048
#define F_DIM 4096
#define NEXP  8
#define XSEL  2
#define A_ASN (T_TOK * XSEL)          // 8192 assignments

// GEMM tiling
#define BM 128
#define BN 64
#define BK 32
#define MAXRB (A_ASN / BM + NEXP)     // 72 worst-case row blocks

#define LDA 40                        // bf16 elems per A-tile row in LDS (32 + 8 pad = 80B, 16B aligned)
#define LDB 40                        // bf16 elems per B-tile column in LDS

typedef __bf16 bf16x16 __attribute__((ext_vector_type(16)));
typedef __bf16 bf16x8  __attribute__((ext_vector_type(8)));
typedef float  f32x8   __attribute__((ext_vector_type(8)));
typedef unsigned int u32x4 __attribute__((ext_vector_type(4)));
typedef int i32x4 __attribute__((ext_vector_type(4)));
typedef int i32x8 __attribute__((ext_vector_type(8)));

union BfBits { unsigned short u; __bf16 b; };

__device__ __forceinline__ __bf16 f2bf(float f) {
    // round-to-nearest-even fp32 -> bf16
    unsigned u = __float_as_uint(f);
    unsigned r = u + 0x7FFFu + ((u >> 16) & 1u);
    BfBits bb; bb.u = (unsigned short)(r >> 16);
    return bb.b;
}

__device__ __forceinline__ bf16x16 ld_frag(const __bf16* p0, const __bf16* p1) {
    union { bf16x16 v; bf16x8 h[2]; } u;
    u.h[0] = *(const bf16x8*)p0;      // 16B aligned ds load
    u.h[1] = *(const bf16x8*)p1;
    return u.v;
}

// ---------------------------------------------------------------------------
// Kernel 1: stable counting sort of assignments by expert + row-block table.
// meta[0]        = num row blocks
// meta[1..9]     = expert offsets (exclusive prefix, 9 entries)
// meta[16..87]   = row-block expert id
// meta[96..167]  = row-block start row (global sorted row)
// ---------------------------------------------------------------------------
__global__ __launch_bounds__(256) void moe_sort_kernel(
    const int* __restrict__ se, int* __restrict__ sorted_token,
    int* __restrict__ pos_of, int* __restrict__ meta) {
    __shared__ int lhist[256][NEXP];
    __shared__ int base[NEXP + 1];
    const int t = threadIdx.x;
    const int CH = A_ASN / 256;       // 32 assignments per thread
    int c[NEXP];
#pragma unroll
    for (int e = 0; e < NEXP; ++e) c[e] = 0;
    const int i0 = t * CH;
    for (int k = 0; k < CH; ++k) c[se[i0 + k]]++;
#pragma unroll
    for (int e = 0; e < NEXP; ++e) lhist[t][e] = c[e];
    __syncthreads();
    if (t < NEXP) {                   // per-expert scan across threads (stable)
        int run = 0;
        for (int th = 0; th < 256; ++th) {
            int v = lhist[th][t];
            lhist[th][t] = run;
            run += v;
        }
        base[t + 1] = run;
    }
    __syncthreads();
    if (t == 0) {
        base[0] = 0;
        for (int e = 0; e < NEXP; ++e) base[e + 1] += base[e];
        for (int e = 0; e <= NEXP; ++e) meta[1 + e] = base[e];
        int nrb = 0;
        for (int e = 0; e < NEXP; ++e)
            for (int s = base[e]; s < base[e + 1]; s += BM) {
                meta[16 + nrb] = e;
                meta[96 + nrb] = s;
                nrb++;
            }
        meta[0] = nrb;
    }
    __syncthreads();
#pragma unroll
    for (int e = 0; e < NEXP; ++e) c[e] = lhist[t][e];
    for (int k = 0; k < CH; ++k) {
        int a = i0 + k;
        int e = se[a];
        int p = base[e] + c[e]++;
        sorted_token[p] = a >> 1;     // a / XSEL
        pos_of[a] = p;
    }
}

// ---------------------------------------------------------------------------
// Kernel 2: fused gating+up grouped GEMM, bf16 WMMA, epilogue silu(g)*u -> h
// ---------------------------------------------------------------------------
__global__ __launch_bounds__(256) void moe_gemm1_kernel(
    const float* __restrict__ x, const float* __restrict__ Wg_all,
    const float* __restrict__ Wu_all, const int* __restrict__ sorted_token,
    const int* __restrict__ meta, __bf16* __restrict__ h) {
    const int nrb = meta[0];
    const int by = blockIdx.y;
    if (by >= nrb) return;
    const int e = meta[16 + by];
    const int rowStart = meta[96 + by];
    const int segEnd = meta[2 + e];   // expert_off[e+1]

    const float* Wg = Wg_all + (size_t)e * D_DIM * F_DIM;
    const float* Wu = Wu_all + (size_t)e * D_DIM * F_DIM;
    const int nBase0 = blockIdx.x * BN;

    __shared__ __bf16 As[BM * LDA];
    __shared__ __bf16 Bg[BN * LDB];   // column-major: Bg[col][k]
    __shared__ __bf16 Bu[BN * LDB];
    __shared__ int toks[BM];

    const int tid = threadIdx.x;
    if (tid < BM) {
        int r = rowStart + tid;
        toks[tid] = sorted_token[r < A_ASN ? r : (A_ASN - 1)];
    }
    __syncthreads();

    const int waveId = tid >> 5;
    const int lane = tid & 31;
    const int waveM = waveId & 3;     // 4 waves over M (4*32 = 128)
    const int waveN = waveId >> 2;    // 2 waves over N (2*32 = 64)
    const int lane16 = lane & 15;
    const int laneH = lane >> 4;

    f32x8 accG[2][2], accU[2][2];
#pragma unroll
    for (int mi = 0; mi < 2; ++mi)
#pragma unroll
        for (int ni = 0; ni < 2; ++ni)
#pragma unroll
            for (int q = 0; q < 8; ++q) { accG[mi][ni][q] = 0.f; accU[mi][ni][q] = 0.f; }

    const int arow = tid >> 3;        // 0..31
    const int akq = (tid & 7) * 4;
    const int bkk = (tid >> 4) * 2;   // 0..30
    const int bnn = (tid & 15) * 4;   // 0..60

    for (int kb = 0; kb < D_DIM / BK; ++kb) {
        const int k0 = kb * BK;
        // --- stage A (gathered x rows, fp32 -> bf16) ---
#pragma unroll
        for (int p = 0; p < 4; ++p) {
            int row = arow + p * 32;
            const float4 v = *(const float4*)(x + (size_t)toks[row] * D_DIM + k0 + akq);
            __bf16* dst = &As[row * LDA + akq];
            dst[0] = f2bf(v.x); dst[1] = f2bf(v.y);
            dst[2] = f2bf(v.z); dst[3] = f2bf(v.w);
        }
        // --- stage Bg/Bu (32x64 tile, transpose to column-major bf16) ---
        {
            const float* g0 = Wg + (size_t)(k0 + bkk) * F_DIM + nBase0 + bnn;
            const float* u0 = Wu + (size_t)(k0 + bkk) * F_DIM + nBase0 + bnn;
            if (kb + 1 < D_DIM / BK) {
                __builtin_prefetch(g0 + (size_t)BK * F_DIM, 0, 1);
                __builtin_prefetch(u0 + (size_t)BK * F_DIM, 0, 1);
            }
            const float4 ga = *(const float4*)g0;
            const float4 gb = *(const float4*)(g0 + F_DIM);
            const float4 ua = *(const float4*)u0;
            const float4 ub = *(const float4*)(u0 + F_DIM);
            const float* pga = &ga.x; const float* pgb = &gb.x;
            const float* pua = &ua.x; const float* pub = &ub.x;
#pragma unroll
            for (int j = 0; j < 4; ++j) {
                Bg[(bnn + j) * LDB + bkk]     = f2bf(pga[j]);
                Bg[(bnn + j) * LDB + bkk + 1] = f2bf(pgb[j]);
                Bu[(bnn + j) * LDB + bkk]     = f2bf(pua[j]);
                Bu[(bnn + j) * LDB + bkk + 1] = f2bf(pub[j]);
            }
        }
        __syncthreads();
        // --- fragments + WMMA ---
        bf16x16 af[2], bgf[2], buf_[2];
#pragma unroll
        for (int mi = 0; mi < 2; ++mi) {
            const __bf16* p = &As[(waveM * 32 + mi * 16 + lane16) * LDA + laneH * 8];
            af[mi] = ld_frag(p, p + 16);
        }
#pragma unroll
        for (int ni = 0; ni < 2; ++ni) {
            const __bf16* pg = &Bg[(waveN * 32 + ni * 16 + lane16) * LDB + laneH * 8];
            bgf[ni] = ld_frag(pg, pg + 16);
            const __bf16* pu = &Bu[(waveN * 32 + ni * 16 + lane16) * LDB + laneH * 8];
            buf_[ni] = ld_frag(pu, pu + 16);
        }
#pragma unroll
        for (int mi = 0; mi < 2; ++mi)
#pragma unroll
            for (int ni = 0; ni < 2; ++ni) {
                accG[mi][ni] = __builtin_amdgcn_wmma_f32_16x16x32_bf16(
                    false, af[mi], false, bgf[ni], (short)0, accG[mi][ni], false, false);
                accU[mi][ni] = __builtin_amdgcn_wmma_f32_16x16x32_bf16(
                    false, af[mi], false, buf_[ni], (short)0, accU[mi][ni], false, false);
            }
        __syncthreads();
    }
    // --- epilogue: h = silu(g) * u (bf16) ---
#pragma unroll
    for (int mi = 0; mi < 2; ++mi)
#pragma unroll
        for (int ni = 0; ni < 2; ++ni)
#pragma unroll
            for (int j = 0; j < 8; ++j) {
                int m = waveM * 32 + mi * 16 + laneH * 8 + j;
                int gr = rowStart + m;
                if (gr < segEnd) {
                    int n = nBase0 + waveN * 32 + ni * 16 + lane16;
                    float g = accG[mi][ni][j];
                    float u = accU[mi][ni][j];
                    float s = g / (1.0f + __expf(-g));
                    h[(size_t)gr * F_DIM + n] = f2bf(s * u);
                }
            }
}

// ---------------------------------------------------------------------------
// Kernel 3: down-projection grouped GEMM (A = h bf16, B = Wdown fp32->bf16)
// A-tile is a pure bf16 global->LDS copy: rows 0..63 staged by the Tensor
// Data Mover (TENSORcnt), rows 64..127 by async global->LDS DMA (ASYNCcnt).
// ---------------------------------------------------------------------------
__global__ __launch_bounds__(256) void moe_gemm2_kernel(
    const __bf16* __restrict__ h, const float* __restrict__ Wd_all,
    const int* __restrict__ meta, float* __restrict__ down) {
    const int nrb = meta[0];
    const int by = blockIdx.y;
    if (by >= nrb) return;
    const int e = meta[16 + by];
    const int rowStart = meta[96 + by];
    const int segEnd = meta[2 + e];

    const float* Wd = Wd_all + (size_t)e * F_DIM * D_DIM;   // [F, D]
    const int nBase0 = blockIdx.x * BN;

    __shared__ __bf16 As[BM * LDA];
    __shared__ __bf16 Bs[BN * LDB];

    const int tid = threadIdx.x;
    const int waveId = tid >> 5;
    const int lane = tid & 31;
    const int waveM = waveId & 3;
    const int waveN = waveId >> 2;
    const int lane16 = lane & 15;
    const int laneH = lane >> 4;

    f32x8 acc[2][2];
#pragma unroll
    for (int mi = 0; mi < 2; ++mi)
#pragma unroll
        for (int ni = 0; ni < 2; ++ni)
#pragma unroll
            for (int q = 0; q < 8; ++q) acc[mi][ni][q] = 0.f;

    const int bkk = (tid >> 4) * 2;
    const int bnn = (tid & 15) * 4;
    const int tdRows = A_ASN - rowStart;     // rows available for TDM OOB check

    for (int kb = 0; kb < F_DIM / BK; ++kb) {
        const int k0 = kb * BK;
        // --- stage A rows 0..63 via Tensor Data Mover (one descriptor, wave 0) ---
#if __has_builtin(__builtin_amdgcn_tensor_load_to_lds)
        if (waveId == 0) {
            unsigned long long gaddr =
                (unsigned long long)(const void*)(h + (size_t)rowStart * F_DIM + k0);
            unsigned ldsbase = (unsigned)(size_t)&As[0];
            u32x4 g0;
            g0[0] = 1u;                                   // count=1, user mode, no gather
            g0[1] = ldsbase;                              // lds_addr
            g0[2] = (unsigned)(gaddr & 0xFFFFFFFFu);      // global_addr[31:0]
            g0[3] = (unsigned)((gaddr >> 32) & 0x01FFFFFFu) | (2u << 30); // addr[56:32] | type=2
            i32x8 g1;
            g1[0] = (1 << 16)      // data_size = 1 (2 bytes)
                  | (1 << 20)      // pad_enable
                  | (3 << 22)      // pad_interval: 16 DWORDs (64B row)
                  | (3 << 25);     // pad_amount: 4 DWORDs (16B pad -> LDA=40 bf16)
            g1[1] = (int)((F_DIM & 0xFFFF) << 16);        // tensor_dim0[15:0]
            g1[2] = (int)(((unsigned)F_DIM >> 16) | (((unsigned)tdRows & 0xFFFFu) << 16));
            g1[3] = (int)((((unsigned)tdRows >> 16) & 0xFFFFu) | (32u << 16)); // tile_dim0=32
            g1[4] = 64;                                   // tile_dim1=64, tile_dim2=0
            g1[5] = F_DIM;                                // tensor_dim0_stride[31:0]
            g1[6] = 0;
            g1[7] = 0;
            i32x4 g2; g2[0] = 0; g2[1] = 0; g2[2] = 0; g2[3] = 0;
            i32x4 g3; g3[0] = 0; g3[1] = 0; g3[2] = 0; g3[3] = 0;
            i32x8 g4; g4[0] = 0; g4[1] = 0; g4[2] = 0; g4[3] = 0;
                      g4[4] = 0; g4[5] = 0; g4[6] = 0; g4[7] = 0;
            __builtin_amdgcn_tensor_load_to_lds(g0, g1, g2, g3, g4, 0);
            __builtin_amdgcn_s_wait_tensorcnt(0);
        }
#else
        if (tid < 128) {                                  // rows 0..63, async DMA fallback
            int row = tid >> 1;
            int half = tid & 1;
            int gr = rowStart + row; if (gr >= A_ASN) gr = A_ASN - 1;
            unsigned long long src =
                (unsigned long long)(const void*)(h + (size_t)gr * F_DIM + k0 + half * 16);
            unsigned l0 = (unsigned)(size_t)&As[row * LDA + half * 16];
            asm volatile("global_load_async_to_lds_b128 %0, %1, off"
                         :: "v"(l0), "v"(src) : "memory");
            asm volatile("global_load_async_to_lds_b128 %0, %1, off offset:16"
                         :: "v"(l0), "v"(src) : "memory");
            asm volatile("s_wait_asynccnt 0x0" ::: "memory");
        }
#endif
        // --- stage A rows 64..127 via async global->LDS DMA (waves 4..7) ---
        if (tid >= 128) {
            int row = 64 + ((tid - 128) >> 1);
            int half = tid & 1;
            int gr = rowStart + row; if (gr >= A_ASN) gr = A_ASN - 1;
            unsigned long long src =
                (unsigned long long)(const void*)(h + (size_t)gr * F_DIM + k0 + half * 16);
            unsigned l0 = (unsigned)(size_t)&As[row * LDA + half * 16];
            asm volatile("global_load_async_to_lds_b128 %0, %1, off"
                         :: "v"(l0), "v"(src) : "memory");
            asm volatile("global_load_async_to_lds_b128 %0, %1, off offset:16"
                         :: "v"(l0), "v"(src) : "memory");
            asm volatile("s_wait_asynccnt 0x0" ::: "memory");
        }
        // --- stage B (32x64 tile of Wd, transpose to column-major bf16) ---
        {
            const float* g0p = Wd + (size_t)(k0 + bkk) * D_DIM + nBase0 + bnn;
            if (kb + 1 < F_DIM / BK)
                __builtin_prefetch(g0p + (size_t)BK * D_DIM, 0, 1);
            const float4 a0 = *(const float4*)g0p;
            const float4 a1 = *(const float4*)(g0p + D_DIM);
            const float* pa0 = &a0.x; const float* pa1 = &a1.x;
#pragma unroll
            for (int j = 0; j < 4; ++j) {
                Bs[(bnn + j) * LDB + bkk]     = f2bf(pa0[j]);
                Bs[(bnn + j) * LDB + bkk + 1] = f2bf(pa1[j]);
            }
        }
        __syncthreads();
        bf16x16 af[2], bf[2];
#pragma unroll
        for (int mi = 0; mi < 2; ++mi) {
            const __bf16* p = &As[(waveM * 32 + mi * 16 + lane16) * LDA + laneH * 8];
            af[mi] = ld_frag(p, p + 16);
        }
#pragma unroll
        for (int ni = 0; ni < 2; ++ni) {
            const __bf16* p = &Bs[(waveN * 32 + ni * 16 + lane16) * LDB + laneH * 8];
            bf[ni] = ld_frag(p, p + 16);
        }
#pragma unroll
        for (int mi = 0; mi < 2; ++mi)
#pragma unroll
            for (int ni = 0; ni < 2; ++ni)
                acc[mi][ni] = __builtin_amdgcn_wmma_f32_16x16x32_bf16(
                    false, af[mi], false, bf[ni], (short)0, acc[mi][ni], false, false);
        __syncthreads();
    }
#pragma unroll
    for (int mi = 0; mi < 2; ++mi)
#pragma unroll
        for (int ni = 0; ni < 2; ++ni)
#pragma unroll
            for (int j = 0; j < 8; ++j) {
                int m = waveM * 32 + mi * 16 + laneH * 8 + j;
                int gr = rowStart + m;
                if (gr < segEnd) {
                    int n = nBase0 + waveN * 32 + ni * 16 + lane16;
                    down[(size_t)gr * D_DIM + n] = acc[mi][ni][j];
                }
            }
}

// ---------------------------------------------------------------------------
// Kernel 4: unpermute + weighted combine: out[t] = sum_x rw[t,x]*down[pos]
// ---------------------------------------------------------------------------
__global__ __launch_bounds__(256) void moe_combine_kernel(
    const float* __restrict__ down, const float* __restrict__ rw,
    const int* __restrict__ pos_of, float* __restrict__ out) {
    const int gid = blockIdx.x * blockDim.x + threadIdx.x;
    const int idx = gid * 4;
    const int t = idx >> 11;          // / D_DIM (2048)
    const int d = idx & (D_DIM - 1);
    const int p0 = pos_of[t * 2 + 0];
    const int p1 = pos_of[t * 2 + 1];
    const float w0 = rw[t * 2 + 0];
    const float w1 = rw[t * 2 + 1];
    const float4 v0 = *(const float4*)(down + (size_t)p0 * D_DIM + d);
    const float4 v1 = *(const float4*)(down + (size_t)p1 * D_DIM + d);
    float4 o;
    o.x = w0 * v0.x + w1 * v1.x;
    o.y = w0 * v0.y + w1 * v1.y;
    o.z = w0 * v0.z + w1 * v1.z;
    o.w = w0 * v0.w + w1 * v1.w;
    *(float4*)(out + idx) = o;
}

// ---------------------------------------------------------------------------
extern "C" void kernel_launch(void* const* d_in, const int* in_sizes, int n_in,
                              void* d_out, int out_size, void* d_ws, size_t ws_size,
                              hipStream_t stream) {
    const float* x  = (const float*)d_in[0];   // [T, D]
    const float* rw = (const float*)d_in[1];   // [T, X]
    const int*   se = (const int*)d_in[2];     // [T, X]
    const float* Wg = (const float*)d_in[3];   // [E, D, F]
    const float* Wu = (const float*)d_in[4];   // [E, D, F]
    const float* Wd = (const float*)d_in[5];   // [E, F, D]
    float* out = (float*)d_out;                // [T, D]

    char* ws = (char*)d_ws;
    int* sorted_token = (int*)(ws);                         // 32 KB
    int* pos_of       = (int*)(ws + 32768);                 // 32 KB
    int* meta         = (int*)(ws + 65536);                 // 1 KB
    __bf16* h         = (__bf16*)(ws + 66560);              // A*F*2 = 64 MB
    float* down       = (float*)(ws + 66560 + (size_t)A_ASN * F_DIM * 2); // 64 MB

    moe_sort_kernel<<<1, 256, 0, stream>>>(se, sorted_token, pos_of, meta);

    dim3 g1(F_DIM / BN, MAXRB);
    moe_gemm1_kernel<<<g1, 256, 0, stream>>>(x, Wg, Wu, sorted_token, meta, h);

    dim3 g2(D_DIM / BN, MAXRB);
    moe_gemm2_kernel<<<g2, 256, 0, stream>>>(h, Wd, meta, down);

    moe_combine_kernel<<<(T_TOK * D_DIM / 4) / 256, 256, 0, stream>>>(down, rw, pos_of, out);
}
